// Block_86311662780900
// MI455X (gfx1250) — matrix operand order repaired and checked by reading, not compiled, not run on hardware
//
#include <hip/hip_runtime.h>

// MI455X / gfx1250, wave32. One wave processes 2 batch elements end-to-end:
//   QKV projections + FFN via v_wmma_f32_16x16x32_bf16 (f32 accum),
//   tiny causal attention via LDS re-layout + packed-f32 (v_pk_fma_f32) math.
// Weights are pre-packed into WMMA B-fragment lane layout (bf16) in d_ws.

typedef __attribute__((ext_vector_type(16))) __bf16 v16bf;
typedef __attribute__((ext_vector_type(8)))  __bf16 v8bf;
typedef __attribute__((ext_vector_type(8)))  float  v8f;
typedef __attribute__((ext_vector_type(4)))  float  v4f;
typedef __attribute__((ext_vector_type(2)))  float  v2f;

static constexpr int T  = 8;
static constexpr int C  = 32;
static constexpr int Dh = 8;

// ---------------------------------------------------------------------------
// Pre-pack Wq/Wk/Wv (H,C,D) and Wf (C,C) into 8 bf16 B-fragments.
// Fragment f, lane l, half j at frag[f*512 + l*16 + j]:
//   B element (kk, n) with n = (l&15)+16*nh, kk = (l>=16 ? 8:0) + j + (j>=8 ? 8:0)
// (matches the CDNA5 16-bit B 32x16 lane striping).
// ---------------------------------------------------------------------------
__global__ void pack_weights_kernel(const float* __restrict__ Wq,
                                    const float* __restrict__ Wk,
                                    const float* __restrict__ Wv,
                                    const float* __restrict__ Wf,
                                    __bf16* __restrict__ frag) {
  const int w    = threadIdx.x >> 5;   // fragment id 0..7
  const int lane = threadIdx.x & 31;
  const int wsel = w >> 1;             // 0=Wq 1=Wk 2=Wv 3=Wf
  const int nh   = w & 1;              // N half
  const float* W = (wsel == 0) ? Wq : (wsel == 1) ? Wk : (wsel == 2) ? Wv : Wf;
  const int n  = (lane & 15) + 16 * nh;
  const int kb = (lane >= 16) ? 8 : 0;
#pragma unroll
  for (int j = 0; j < 16; ++j) {
    const int kk = kb + j + (j >= 8 ? 8 : 0);
    float v;
    if (wsel < 3) v = W[(n >> 3) * (C * Dh) + kk * Dh + (n & 7)]; // col n = h*8+d
    else          v = W[kk * C + n];
    frag[w * 512 + lane * 16 + j] = (__bf16)v;
  }
}

// ---------------------------------------------------------------------------
// Main fused kernel: 256 threads = 8 waves; each wave owns 2 batch elements.
// ---------------------------------------------------------------------------
__global__ void __launch_bounds__(256)
attn_block_kernel(const float* __restrict__ x,
                  const __bf16* __restrict__ wfrag,
                  const float* __restrict__ bias,
                  float* __restrict__ out) {
  // per-wave LDS: q/k/v staging (f32, 72-float pitch per (b,h) block to avoid
  // bank conflicts) + attention output staged as bf16 in A-fragment order.
  __shared__ float  qkv[8][3][8 * 72];     // 55296 B
  __shared__ __bf16 obuf[8][16][32];       //  8192 B  (total 63488 B < 64 KB)

  const int tid  = threadIdx.x;
  const int w    = tid >> 5;
  const int lane = tid & 31;
  const long long b0 = ((long long)blockIdx.x * 8 + w) * 2;

  // ---- weight B-fragments (L2-resident, 2x b128 per lane each) ----
  const v16bf* fr = (const v16bf*)wfrag;
  const v16bf wq0 = fr[0 * 32 + lane], wq1 = fr[1 * 32 + lane];
  const v16bf wk0 = fr[2 * 32 + lane], wk1 = fr[3 * 32 + lane];
  const v16bf wv0 = fr[4 * 32 + lane], wv1 = fr[5 * 32 + lane];
  const v16bf wf0 = fr[6 * 32 + lane], wf1 = fr[7 * 32 + lane];

  // ---- build A fragment from x: M = 2 batches x 8 t, K = C = 32 ----
  // lane layout: M = lane&15, halves 0-7 -> K = Kb..Kb+7, 8-15 -> 16+Kb..,
  // Kb = (lane>=16)*8.
  const int m    = lane & 15;
  const int bsel = m >> 3;
  const int tA   = m & 7;
  const int kbA  = (lane >= 16) ? 8 : 0;
  const float* xrow = x + ((b0 + bsel) * T + tA) * C;
  const v4f x0 = *(const v4f*)(xrow + kbA);
  const v4f x1 = *(const v4f*)(xrow + kbA + 4);
  const v4f x2 = *(const v4f*)(xrow + 16 + kbA);
  const v4f x3 = *(const v4f*)(xrow + 16 + kbA + 4);
  v16bf ax;
#pragma unroll
  for (int j = 0; j < 4; ++j) {
    ax[j]      = (__bf16)x0[j];
    ax[4 + j]  = (__bf16)x1[j];
    ax[8 + j]  = (__bf16)x2[j];
    ax[12 + j] = (__bf16)x3[j];
  }

  // ---- QKV projections: 6 x v_wmma_f32_16x16x32_bf16 ----
  v8f z = {};
  v8f dq0 = __builtin_amdgcn_wmma_f32_16x16x32_bf16(false, ax, false, wq0, (short)0, z, false, false);
  v8f dq1 = __builtin_amdgcn_wmma_f32_16x16x32_bf16(false, ax, false, wq1, (short)0, z, false, false);
  v8f dk0 = __builtin_amdgcn_wmma_f32_16x16x32_bf16(false, ax, false, wk0, (short)0, z, false, false);
  v8f dk1 = __builtin_amdgcn_wmma_f32_16x16x32_bf16(false, ax, false, wk1, (short)0, z, false, false);
  v8f dv0 = __builtin_amdgcn_wmma_f32_16x16x32_bf16(false, ax, false, wv0, (short)0, z, false, false);
  v8f dv1 = __builtin_amdgcn_wmma_f32_16x16x32_bf16(false, ax, false, wv1, (short)0, z, false, false);

  // ---- stage q/k/v to LDS. D layout: lane holds (b=(lane>=16), h, d=lane&7),
  // register r = t.  Block g = b*4+h, pitch 72 floats (conflict-free). ----
  {
    const int d  = lane & 7;
    const int bb = lane >> 4;
    const int h0 = (lane >> 3) & 1;
    const int g0 = bb * 4 + h0;       // frag0: heads 0/1
    const int g1 = bb * 4 + 2 + h0;   // frag1: heads 2/3
#pragma unroll
    for (int r = 0; r < 8; ++r) {
      qkv[w][0][g0 * 72 + r * 8 + d] = dq0[r];
      qkv[w][0][g1 * 72 + r * 8 + d] = dq1[r];
      qkv[w][1][g0 * 72 + r * 8 + d] = dk0[r];
      qkv[w][1][g1 * 72 + r * 8 + d] = dk1[r];
      qkv[w][2][g0 * 72 + r * 8 + d] = dv0[r];
      qkv[w][2][g1 * 72 + r * 8 + d] = dv1[r];
    }
  }
  __syncthreads();

  // ---- causal attention: lane -> (b, h, pair of t); packed-f32 math ----
  {
    const int ab  = lane >> 4;        // local batch
    const int ah  = (lane >> 2) & 3;  // head
    const int atp = lane & 3;         // t pair
    const int g   = ab * 4 + ah;
    const float* qb = &qkv[w][0][g * 72];
    const float* kb = &qkv[w][1][g * 72];
    const float* vb = &qkv[w][2][g * 72];
    // scale folded with log2(e): exp(x*s) = exp2(x * s * log2e)
    const float scale2 = 0.35355339059327373f * 1.4426950408889634f;

    float wei[2][8];
    float winv[2];
    {
      v2f kreg[8][4];
#pragma unroll
      for (int s = 0; s < 8; ++s) {
        const v4f ka = *(const v4f*)(kb + s * 8);
        const v4f kc = *(const v4f*)(kb + s * 8 + 4);
        kreg[s][0] = (v2f){ka[0], ka[1]};
        kreg[s][1] = (v2f){ka[2], ka[3]};
        kreg[s][2] = (v2f){kc[0], kc[1]};
        kreg[s][3] = (v2f){kc[2], kc[3]};
      }
#pragma unroll
      for (int t2 = 0; t2 < 2; ++t2) {
        const int tt = atp * 2 + t2;
        const v4f qa = *(const v4f*)(qb + tt * 8);
        const v4f qc = *(const v4f*)(qb + tt * 8 + 4);
        const v2f qs = (v2f){scale2, scale2};
        v2f qv[4];
        qv[0] = (v2f){qa[0], qa[1]} * qs;   // pre-scale q: v_pk_mul_f32
        qv[1] = (v2f){qa[2], qa[3]} * qs;
        qv[2] = (v2f){qc[0], qc[1]} * qs;
        qv[3] = (v2f){qc[2], qc[3]} * qs;
        float sc[8];
#pragma unroll
        for (int s = 0; s < 8; ++s) {
          v2f acc = qv[0] * kreg[s][0];     // v_pk_mul / v_pk_fma chain
          acc += qv[1] * kreg[s][1];
          acc += qv[2] * kreg[s][2];
          acc += qv[3] * kreg[s][3];
          const float d8 = acc[0] + acc[1];
          sc[s] = (s <= tt) ? d8 : -1e30f;  // causal mask (log2-domain)
        }
        float mx = sc[0];
#pragma unroll
        for (int s = 1; s < 8; ++s) mx = fmaxf(mx, sc[s]);
        float sum = 0.f;
#pragma unroll
        for (int s = 0; s < 8; ++s) {
          wei[t2][s] = __builtin_amdgcn_exp2f(sc[s] - mx);  // v_exp_f32
          sum += wei[t2][s];
        }
        winv[t2] = __builtin_amdgcn_rcpf(sum);              // v_rcp_f32
      }
    }
    // out = (wei @ v) * winv -> obuf[b*8+t][h*8+d] (bf16, A-fragment order)
    {
      v2f vreg[8][4];
#pragma unroll
      for (int s = 0; s < 8; ++s) {
        const v4f va = *(const v4f*)(vb + s * 8);
        const v4f vc = *(const v4f*)(vb + s * 8 + 4);
        vreg[s][0] = (v2f){va[0], va[1]};
        vreg[s][1] = (v2f){va[2], va[3]};
        vreg[s][2] = (v2f){vc[0], vc[1]};
        vreg[s][3] = (v2f){vc[2], vc[3]};
      }
#pragma unroll
      for (int t2 = 0; t2 < 2; ++t2) {
        const int tt = atp * 2 + t2;
        v2f ov[4] = {{0.f, 0.f}, {0.f, 0.f}, {0.f, 0.f}, {0.f, 0.f}};
#pragma unroll
        for (int s = 0; s < 8; ++s) {
          const v2f ws2 = (v2f){wei[t2][s], wei[t2][s]};
#pragma unroll
          for (int p = 0; p < 4; ++p) ov[p] += ws2 * vreg[s][p];  // v_pk_fma_f32
        }
        const v2f iv2 = (v2f){winv[t2], winv[t2]};
        v8bf ob;
#pragma unroll
        for (int p = 0; p < 4; ++p) {
          const v2f o2 = ov[p] * iv2;       // fold softmax normalization
          ob[2 * p]     = (__bf16)o2[0];
          ob[2 * p + 1] = (__bf16)o2[1];
        }
        *(v8bf*)&obuf[w][ab * 8 + tt][ah * 8] = ob;   // ds_store_b128
      }
    }
  }
  __syncthreads();

  // ---- FFN: A from obuf (already in fragment order), 2 WMMAs, bias+ReLU ----
  {
    const int row = (lane & 15);           // M = b'*8 + t
    const int kb2 = (lane >= 16) ? 8 : 0;  // K base
    const v8bf a0 = *(const v8bf*)&obuf[w][row][kb2];
    const v8bf a1 = *(const v8bf*)&obuf[w][row][16 + kb2];
    v16bf af;
#pragma unroll
    for (int j = 0; j < 8; ++j) { af[j] = a0[j]; af[8 + j] = a1[j]; }

    v8f df0 = __builtin_amdgcn_wmma_f32_16x16x32_bf16(false, af, false, wf0, (short)0, z, false, false);
    v8f df1 = __builtin_amdgcn_wmma_f32_16x16x32_bf16(false, af, false, wf1, (short)0, z, false, false);

    const float bi0 = bias[lane & 15];
    const float bi1 = bias[(lane & 15) + 16];
    float* orow = out + (b0 + (lane >= 16)) * (T * C);
#pragma unroll
    for (int r = 0; r < 8; ++r) {
      orow[r * C + (lane & 15)]      = fmaxf(df0[r] + bi0, 0.f);
      orow[r * C + (lane & 15) + 16] = fmaxf(df1[r] + bi1, 0.f);
    }
  }
}

// ---------------------------------------------------------------------------
extern "C" void kernel_launch(void* const* d_in, const int* in_sizes, int n_in,
                              void* d_out, int out_size, void* d_ws, size_t ws_size,
                              hipStream_t stream) {
  const float* x  = (const float*)d_in[0];
  const float* Wq = (const float*)d_in[1];
  const float* Wk = (const float*)d_in[2];
  const float* Wv = (const float*)d_in[3];
  const float* Wf = (const float*)d_in[4];
  const float* bf = (const float*)d_in[5];
  float* outp = (float*)d_out;

  __bf16* frag = (__bf16*)d_ws;  // needs 8 KB of workspace

  const int nb = in_sizes[0] / (T * C);     // 131072 batches
  const int blocks = nb / 16;               // 8 waves/block * 2 batches/wave

  pack_weights_kernel<<<1, 256, 0, stream>>>(Wq, Wk, Wv, Wf, frag);
  attn_block_kernel<<<blocks, 256, 0, stream>>>(x, frag, bf, outp);
}
